// GumbelVectorQuantizer_3839700763052
// MI455X (gfx1250) — matrix operand order repaired and checked by reading, not compiled, not run on hardware
//
#include <hip/hip_runtime.h>
#include <hip/hip_bf16.h>

#define B_   16
#define T_   3000
#define D_   512
#define GV_  640        // GROUPS*NUM_VARS
#define V_   320        // NUM_VARS
#define VD_  128        // VAR_DIM
#define N_   (B_*T_)    // 48000 tokens

typedef __attribute__((ext_vector_type(16))) __bf16 v16bf;
typedef __attribute__((ext_vector_type(8)))  float  v8f;

union Frag { v16bf v; unsigned int u[8]; };

__device__ __forceinline__ unsigned short f32_bf16(float f) {
    unsigned int u = __float_as_uint(f);
    u += 0x7FFFu + ((u >> 16) & 1u);          // round to nearest even
    return (unsigned short)(u >> 16);
}
__device__ __forceinline__ unsigned int pack_bf16(float lo, float hi) {
    return (unsigned int)f32_bf16(lo) | ((unsigned int)f32_bf16(hi) << 16);
}

// ---- Kernel 0: W fp32 -> bf16 (packed pairs along K), zero avg accumulator ----
__global__ __launch_bounds__(256) void prep_kernel(const float* __restrict__ W,
                                                   unsigned int* __restrict__ Wbf,
                                                   float* __restrict__ acc) {
    int i = blockIdx.x * 256 + threadIdx.x;   // pair index, total GV_*D_/2 = 163840
    if (i < GV_ * D_ / 2) Wbf[i] = pack_bf16(W[2 * i], W[2 * i + 1]);
    if (i < GV_) acc[i] = 0.0f;
}

// ---- Kernel 1: fused GEMM(bf16 WMMA) + argmax + softmax-accumulate + gather ----
__global__ __launch_bounds__(256) void vq_kernel(const float* __restrict__ x,
                                                 const unsigned int* __restrict__ Wbf,
                                                 const float* __restrict__ bias,
                                                 const float* __restrict__ cbook,
                                                 float* __restrict__ out,
                                                 float* __restrict__ acc) {
    __shared__ unsigned int sx[16 * 256];     // 16 tokens x 512 K as bf16 pairs (16 KB)
    __shared__ float slog[16 * GV_];          // logits tile (40 KB)
    __shared__ float sred[256];
    __shared__ int   sredi[256];
    __shared__ float srmax[32];
    __shared__ int   sridx[32];
    __shared__ float srsum[32];
    __shared__ float sprob[GV_];

    const int tid  = threadIdx.x;
    const int tok0 = blockIdx.x * 16;

    // Stage + convert x tile: 4096 bf16 pairs
    for (int i = tid; i < 16 * 256; i += 256) {
        int row = i >> 8, p = i & 255;
        const float* src = x + (size_t)(tok0 + row) * D_ + 2 * p;
        sx[i] = pack_bf16(src[0], src[1]);
    }
    for (int i = tid; i < GV_; i += 256) sprob[i] = 0.0f;
    __syncthreads();

    const int lane = tid & 31;
    const int wave = tid >> 5;
    const int half = lane >> 4;               // K-half select (contiguous k = 16*half + s in this code)
    const int ln   = lane & 15;               // row (A) / col (B)
    const int cwave = wave * 80;              // each wave owns 80 columns (5 tiles)

    // Per-lane base pointers
    const unsigned int* arow = &sx[(ln << 8) + (half << 3)];
    const unsigned int* brow[5];
#pragma unroll
    for (int t = 0; t < 5; ++t)
        brow[t] = Wbf + (size_t)(cwave + t * 16 + ln) * (D_ / 2) + (half << 3);

    v8f  accv[5] = {};
    Frag bcur[5], bnxt[5];

    // Prologue: load B fragments for K-step 0 (10x b128, one clause)
#pragma unroll
    for (int t = 0; t < 5; ++t) {
#pragma unroll
        for (int j = 0; j < 8; ++j) bcur[t].u[j] = brow[t][j];
    }

#pragma unroll
    for (int ks = 0; ks < 16; ++ks) {         // K = 512 in steps of 32
        const int kb = ks * 16;               // index in bf16-pair units

        // A 16x32 bf16: lanes 0-15 hold K 0-15; lanes 16-31 hold K 16-31
        Frag a;
#pragma unroll
        for (int j = 0; j < 4; ++j) a.u[j]     = arow[kb + j];
#pragma unroll
        for (int j = 0; j < 4; ++j) a.u[4 + j] = arow[kb + 4 + j];

        // Prefetch B fragments for K-step ks+1 (double buffer; hides load latency
        // behind the 5 WMMAs of this step)
        if (ks < 15) {
            const int kn = kb + 16;
#pragma unroll
            for (int t = 0; t < 5; ++t) {
#pragma unroll
                for (int j = 0; j < 8; ++j) bnxt[t].u[j] = brow[t][kn + j];
            }
        }

#pragma unroll
        for (int t = 0; t < 5; ++t)
            accv[t] = __builtin_amdgcn_wmma_f32_16x16x32_bf16(
                false, a.v, false, bcur[t].v, (short)0, accv[t], false, false);

#pragma unroll
        for (int t = 0; t < 5; ++t) bcur[t] = bnxt[t];   // register rotation
    }

    // C layout: VGPR r holds row M=r (lanes<16) / M=r+8 (lanes>=16), col N = lane&15
#pragma unroll
    for (int t = 0; t < 5; ++t) {
        int col = cwave + t * 16 + ln;
        float bv = bias[col];
#pragma unroll
        for (int r = 0; r < 8; ++r)
            slog[(r + 8 * half) * GV_ + col] = accv[t][r] + bv;
    }
    __syncthreads();

    // 32 tasks (row 0..15 x group 0..1), 8 subthreads each scanning 40 cols
    const int task = tid >> 3;
    const int sub  = tid & 7;
    const int row  = task >> 1;
    const int grp  = task & 1;
    const float* lrow = &slog[row * GV_ + grp * V_];

    // Phase A: argmax (first occurrence wins)
    float m = -3.4e38f; int mi = V_;
    for (int c = sub * 40; c < sub * 40 + 40; ++c) {
        float v = lrow[c];
        if (v > m) { m = v; mi = c; }
    }
    sred[tid] = m; sredi[tid] = mi;
    __syncthreads();
    if (tid < 32) {
        float bm = sred[tid * 8]; int bi = sredi[tid * 8];
        for (int j = 1; j < 8; ++j) {
            float v = sred[tid * 8 + j]; int vi = sredi[tid * 8 + j];
            if (v > bm || (v == bm && vi < bi)) { bm = v; bi = vi; }
        }
        srmax[tid] = bm; sridx[tid] = bi;
    }
    __syncthreads();

    // Phase B: sum of exp (stable)
    float rmax = srmax[task];
    float s = 0.0f;
    for (int c = sub * 40; c < sub * 40 + 40; ++c) s += __expf(lrow[c] - rmax);
    sred[tid] = s;
    __syncthreads();
    if (tid < 32) {
        float tot = 0.0f;
        for (int j = 0; j < 8; ++j) tot += sred[tid * 8 + j];
        srsum[tid] = tot;
    }
    __syncthreads();

    // Phase C: per-column probability accumulation (over 16 rows) into LDS
    float rinv = 1.0f / srsum[task];
    for (int c = sub * 40; c < sub * 40 + 40; ++c) {
        float p = __expf(lrow[c] - rmax) * rinv;
        atomicAdd(&sprob[grp * V_ + c], p);
    }
    __syncthreads();

    // Phase D: one pass of global accumulation (640 atomics per workgroup)
    for (int i = tid; i < GV_; i += 256) atomicAdd(&acc[i], sprob[i]);

    // Gather selected code vectors: 32 tasks x 128 floats, coalesced writes
    for (int i = tid; i < 16 * 2 * VD_; i += 256) {
        int d  = i & (VD_ - 1);
        int tk = i >> 7;                 // row*2 + g
        int r2 = tk >> 1, g2 = tk & 1;
        int idx = sridx[r2 * 2 + g2];
        out[(size_t)(tok0 + r2) * (2 * VD_) + g2 * VD_ + d] =
            cbook[((size_t)g2 * V_ + idx) * VD_ + d];
    }
}

// ---- Kernel 2: finalize avg_probs ----
__global__ __launch_bounds__(256) void finalize_kernel(const float* __restrict__ acc,
                                                       float* __restrict__ out_avg) {
    int i = blockIdx.x * 256 + threadIdx.x;
    if (i < GV_) out_avg[i] = acc[i] * (1.0f / (float)N_);
}

extern "C" void kernel_launch(void* const* d_in, const int* in_sizes, int n_in,
                              void* d_out, int out_size, void* d_ws, size_t ws_size,
                              hipStream_t stream) {
    const float* x  = (const float*)d_in[0];   // [16,3000,512]
    const float* W  = (const float*)d_in[1];   // [640,512]
    const float* b  = (const float*)d_in[2];   // [640]
    const float* cb = (const float*)d_in[3];   // [1,640,128]
    float* out = (float*)d_out;                // [48000*256] quantized + [640] avg

    float* acc = (float*)d_ws;                                   // 640 floats
    unsigned int* Wbf = (unsigned int*)((char*)d_ws + 4096);     // 640 KB bf16 W

    prep_kernel<<<(GV_ * D_ / 2 + 255) / 256, 256, 0, stream>>>(W, Wbf, acc);
    vq_kernel<<<N_ / 16, 256, 0, stream>>>(x, Wbf, b, cb, out, acc);
    finalize_kernel<<<(GV_ + 255) / 256, 256, 0, stream>>>(acc, out + (size_t)N_ * 2 * VD_);
}
